// Attention_3358664425682
// MI455X (gfx1250) — compile-verified, matrix operand order checked by reading, not run on hardware
//
#include <hip/hip_runtime.h>
#include <hip/hip_bf16.h>
#include <math.h>

// ---------------------------------------------------------------------------
// Gated self-attention + LayerNorm for MI455X (gfx1250), bf16 WMMA path.
// B=4, N=2048, DIM=512, HEADS=8, DIM_HEAD=64
// ---------------------------------------------------------------------------

#define DIMC     512
#define HEADS    8
#define DHEAD    64
#define BATCH    4
#define SEQ      2048
#define ROWS     (BATCH * SEQ)        // 8192
#define LN_EPS   1e-5f

typedef __attribute__((ext_vector_type(16))) __bf16         bf16x16;
typedef __attribute__((ext_vector_type(8)))  float          floatx8;
typedef __attribute__((ext_vector_type(8)))  unsigned short u16x8;
typedef __attribute__((ext_vector_type(16))) unsigned short u16x16;
typedef __attribute__((ext_vector_type(4)))  int            intx4;

__device__ __forceinline__ unsigned short f32_to_bf16(float f) {
    unsigned int u = __float_as_uint(f);
    u += 0x7FFFu + ((u >> 16) & 1u);          // round-to-nearest-even
    return (unsigned short)(u >> 16);
}

// ---------------------------------------------------------------------------
// Async global->LDS copy (16 bytes / lane), ASYNCcnt-tracked on gfx1250.
// Builtin signature (from clang diagnostic): param0 = addrspace(1) int4*,
// param1 = LDS int4*, then two integer immediates (offset, cpol).
// LDS flat addresses truncate to the 32-bit LDS offset (aperture lives in
// the upper 32 bits), so integer round-trips give valid AS(1)/AS(3) ptrs.
// ---------------------------------------------------------------------------
#if __has_builtin(__builtin_amdgcn_global_load_async_to_lds_b128)
#define HAVE_ASYNC_LDS 1
#else
#define HAVE_ASYNC_LDS 0
#endif

__device__ __forceinline__ void async_copy16(const unsigned short* g,
                                             unsigned short* l) {
#if HAVE_ASYNC_LDS
    __builtin_amdgcn_global_load_async_to_lds_b128(
        (__attribute__((address_space(1))) intx4*)(unsigned long long)(size_t)g,
        (__attribute__((address_space(3))) intx4*)(unsigned int)(size_t)l,
        0, 0);
#else
    *(u16x8*)l = *(const u16x8*)g;
#endif
}

__device__ __forceinline__ void async_join() {
#if HAVE_ASYNC_LDS
#if __has_builtin(__builtin_amdgcn_s_wait_asynccnt)
    __builtin_amdgcn_s_wait_asynccnt(0);
#else
    asm volatile("s_wait_asynccnt 0" ::: "memory");
#endif
#endif
}

// Load one 16-bf16 WMMA fragment (A-layout / transposed-B layout).
// Per the CDNA5 ISA 16-bit A layout: lane half selects K-runs
//   elements 0..7  -> K = kc + half*8 .. +7        (contiguous, 16B)
//   elements 8..15 -> K = kc + 16 + half*8 .. +7   (contiguous, 16B)
// so a fragment is exactly two aligned 16-byte loads.
__device__ __forceinline__ bf16x16 load_frag(const unsigned short* base,
                                             int rowstride, int row,
                                             int kc, int half) {
    const unsigned short* p = base + row * rowstride + kc + half * 8;
    u16x8 lo = *(const u16x8*)(p);
    u16x8 hi = *(const u16x8*)(p + 16);
    u16x16 v = __builtin_shufflevector(lo, hi, 0,1,2,3,4,5,6,7,
                                               8,9,10,11,12,13,14,15);
    return __builtin_bit_cast(bf16x16, v);
}

__device__ __forceinline__ floatx8 wmma_bf16(bf16x16 a, bf16x16 b, floatx8 c) {
    return __builtin_amdgcn_wmma_f32_16x16x32_bf16(
        /*neg_a=*/false, a, /*neg_b=*/false, b,
        /*c_mod=*/(short)0, c, /*reuse_a=*/false, /*reuse_b=*/false);
}

// ---------------------------------------------------------------------------
// fp32 -> bf16 (stored as u16) conversion
// ---------------------------------------------------------------------------
__global__ void k_cvt_bf16(const float* __restrict__ src,
                           unsigned short* __restrict__ dst, int n) {
    int i = blockIdx.x * blockDim.x + threadIdx.x;
    if (i < n) dst[i] = f32_to_bf16(src[i]);
}

// ---------------------------------------------------------------------------
// Generic bf16 GEMM: C[M,N] = A[M,K] @ B[K,N]  (row-major, fp32 accum)
// 64x64 tile per 128-thread workgroup; each wave owns a 16x64 stripe.
// K-step 64: A staged by async b128 copies, B staged transposed ([n][k]).
// Optional fp32 output (Cf), optional bf16 output (Cb), bias + scale.
// ---------------------------------------------------------------------------
__global__ __launch_bounds__(128)
void k_gemm_bf16(const unsigned short* __restrict__ A,
                 const unsigned short* __restrict__ B,
                 int M, int N, int K,
                 const float* __restrict__ bias, float scale,
                 float* __restrict__ Cf, unsigned short* __restrict__ Cb) {
    __shared__ unsigned short As[64 * 64];   // [m][k]
    __shared__ unsigned short Bs[64 * 64];   // [n][k]  (transposed tile)

    const int tid  = threadIdx.x;
    const int wave = tid >> 5;
    const int lane = tid & 31;
    const int half = lane >> 4;
    const int ln   = lane & 15;
    const int m0   = blockIdx.y * 64;
    const int n0   = blockIdx.x * 64;

    // Staging ownership: 128 threads x 32 contiguous elements = 64x64 tile.
    const int srow = tid >> 1;               // 0..63
    const int scol = (tid & 1) * 32;         // 0 or 32

    floatx8 acc[4];
#pragma unroll
    for (int t = 0; t < 4; ++t) acc[t] = floatx8{0,0,0,0,0,0,0,0};

    for (int kc = 0; kc < K; kc += 64) {
        __syncthreads();
        // A tile: contiguous -> contiguous, async b128 x4 per thread.
        {
            const unsigned short* g = A + (size_t)(m0 + srow) * K + kc + scol;
            unsigned short*       l = As + srow * 64 + scol;
#pragma unroll
            for (int j = 0; j < 4; ++j) async_copy16(g + j * 8, l + j * 8);
        }
        // B tile transposed: scalar global gather (stride N), b128 DS stores.
        {
            const int n = srow;              // output column within tile
#pragma unroll
            for (int v = 0; v < 4; ++v) {
                u16x8 pk;
#pragma unroll
                for (int j = 0; j < 8; ++j)
                    pk[j] = B[(size_t)(kc + scol + v * 8 + j) * N + n0 + n];
                *(u16x8*)(Bs + n * 64 + scol + v * 8) = pk;
            }
        }
        async_join();
        __syncthreads();

        bf16x16 a0 = load_frag(As, 64, wave * 16 + ln, 0,  half);
        bf16x16 a1 = load_frag(As, 64, wave * 16 + ln, 32, half);
#pragma unroll
        for (int t = 0; t < 4; ++t) {
            bf16x16 b0 = load_frag(Bs, 64, t * 16 + ln, 0,  half);
            bf16x16 b1 = load_frag(Bs, 64, t * 16 + ln, 32, half);
            acc[t] = wmma_bf16(a0, b0, acc[t]);
            acc[t] = wmma_bf16(a1, b1, acc[t]);
        }
    }

#pragma unroll
    for (int t = 0; t < 4; ++t) {
#pragma unroll
        for (int r = 0; r < 8; ++r) {
            int m   = r + 8 * half;                  // C layout: VGPR r, half
            int row = m0 + wave * 16 + m;
            int col = n0 + t * 16 + ln;
            float v = acc[t][r] * scale + (bias ? bias[col] : 0.0f);
            if (Cf) Cf[(size_t)row * N + col] = v;
            if (Cb) Cb[(size_t)row * N + col] = f32_to_bf16(v);
        }
    }
}

// ---------------------------------------------------------------------------
// Flash attention: one workgroup = 64 query rows of one (b, h).
// 4 waves; each wave owns 16 query rows, O accum = 4 x (16x16) fp32 tiles.
// K-block = 64 keys per iteration; online softmax; scores never hit HBM.
// K tile staged via async b128 copies; V tile staged transposed.
// q is pre-scaled by d^-0.5 at projection time.
// ---------------------------------------------------------------------------
__global__ __launch_bounds__(128)
void k_flash_attn(const unsigned short* __restrict__ qb,   // [ROWS, 512] bf16
                  const unsigned short* __restrict__ kvb,  // [ROWS, 1024] bf16
                  float* __restrict__ attn_out) {          // [ROWS, 512] f32
    __shared__ unsigned short Ks[64 * 64];     // [key][dim]
    __shared__ unsigned short Vt[64 * 64];     // [dim][key] (transposed)
    __shared__ unsigned short Ps[4][16 * 64];  // per wave: [m][key], bf16

    const int tid  = threadIdx.x;
    const int wave = tid >> 5;
    const int lane = tid & 31;
    const int half = lane >> 4;
    const int ln   = lane & 15;

    const int qblk = blockIdx.x;               // 0..31
    const int bh   = blockIdx.y;               // 0..31
    const int b    = bh / HEADS;
    const int h    = bh % HEADS;

    const int srow = tid >> 1;                 // 0..63
    const int scol = (tid & 1) * 32;           // 0 or 32

    // Load Q fragments once (A layout, row = lane%16, strided from global).
    const int qrow = qblk * 64 + wave * 16 + ln;
    const unsigned short* qbase = qb + (size_t)(b * SEQ + qrow) * DIMC + h * DHEAD;
    bf16x16 aq0 = load_frag(qbase, 0, 0, 0,  half);
    bf16x16 aq1 = load_frag(qbase, 0, 0, 32, half);

    floatx8 o[4];
#pragma unroll
    for (int t = 0; t < 4; ++t) o[t] = floatx8{0,0,0,0,0,0,0,0};
    float mrow[8], lrow[8];
#pragma unroll
    for (int r = 0; r < 8; ++r) { mrow[r] = -3.0e38f; lrow[r] = 0.0f; }

    for (int jb = 0; jb < SEQ / 64; ++jb) {
        __syncthreads();                       // Ks/Vt safe to overwrite
        // K tile [key][dim]: contiguous rows -> async b128 copies.
        {
            const unsigned short* g =
                kvb + (size_t)(b * SEQ + jb * 64 + srow) * (2 * DIMC) + h * DHEAD + scol;
            unsigned short* l = Ks + srow * 64 + scol;
#pragma unroll
            for (int j = 0; j < 4; ++j) async_copy16(g + j * 8, l + j * 8);
        }
        // V tile transposed [dim][key]: scalar gather, b128 DS stores.
        {
            const int d = srow;                // dim within head
#pragma unroll
            for (int v = 0; v < 4; ++v) {
                u16x8 pk;
#pragma unroll
                for (int j = 0; j < 8; ++j)
                    pk[j] = kvb[(size_t)(b * SEQ + jb * 64 + scol + v * 8 + j) * (2 * DIMC)
                                + DIMC + h * DHEAD + d];
                *(u16x8*)(Vt + d * 64 + scol + v * 8) = pk;
            }
        }
        async_join();
        __syncthreads();

        // S = Q @ K^T   (B fragment: n = key (fixed/lane), k = dim (contig))
        floatx8 s[4];
#pragma unroll
        for (int t = 0; t < 4; ++t) {
            s[t] = floatx8{0,0,0,0,0,0,0,0};
            bf16x16 bk0 = load_frag(Ks, 64, t * 16 + ln, 0,  half);
            bf16x16 bk1 = load_frag(Ks, 64, t * 16 + ln, 32, half);
            s[t] = wmma_bf16(aq0, bk0, s[t]);
            s[t] = wmma_bf16(aq1, bk1, s[t]);
        }

        // Online softmax. Row r+8*half is shared by the 16 lanes of `half`,
        // so xor-reductions over masks {1,2,4,8} stay inside the row group.
#pragma unroll
        for (int r = 0; r < 8; ++r) {
            float mx = s[0][r];
#pragma unroll
            for (int t = 1; t < 4; ++t) mx = fmaxf(mx, s[t][r]);
#pragma unroll
            for (int msk = 1; msk <= 8; msk <<= 1)
                mx = fmaxf(mx, __shfl_xor(mx, msk, 32));
            float mnew  = fmaxf(mrow[r], mx);
            float alpha = __expf(mrow[r] - mnew);
            float psum  = 0.0f;
#pragma unroll
            for (int t = 0; t < 4; ++t) {
                float p = __expf(s[t][r] - mnew);
                s[t][r] = p;
                psum += p;
            }
#pragma unroll
            for (int msk = 1; msk <= 8; msk <<= 1)
                psum += __shfl_xor(psum, msk, 32);
            lrow[r] = lrow[r] * alpha + psum;
            mrow[r] = mnew;
#pragma unroll
            for (int t = 0; t < 4; ++t) o[t][r] *= alpha;
        }

        // P (C layout) -> per-wave LDS as bf16 in A layout ([m][key]).
#pragma unroll
        for (int t = 0; t < 4; ++t)
#pragma unroll
            for (int r = 0; r < 8; ++r)
                Ps[wave][(r + 8 * half) * 64 + t * 16 + ln] = f32_to_bf16(s[t][r]);
        __syncthreads();

        // O += P @ V   (A from Ps, B from transposed V tile)
        bf16x16 ap0 = load_frag(Ps[wave], 64, ln, 0,  half);
        bf16x16 ap1 = load_frag(Ps[wave], 64, ln, 32, half);
#pragma unroll
        for (int t = 0; t < 4; ++t) {
            bf16x16 bv0 = load_frag(Vt, 64, t * 16 + ln, 0,  half);
            bf16x16 bv1 = load_frag(Vt, 64, t * 16 + ln, 32, half);
            o[t] = wmma_bf16(ap0, bv0, o[t]);
            o[t] = wmma_bf16(ap1, bv1, o[t]);
        }
    }

    // Normalize and write O  (head-major [b, n, h*d] layout).
#pragma unroll
    for (int t = 0; t < 4; ++t)
#pragma unroll
        for (int r = 0; r < 8; ++r) {
            int m   = r + 8 * half;
            int row = qblk * 64 + wave * 16 + m;
            float v = o[t][r] / lrow[r];
            attn_out[(size_t)(b * SEQ + row) * DIMC + h * DHEAD + t * 16 + ln] = v;
        }
}

// ---------------------------------------------------------------------------
// gated = attn * sigmoid(gates)  -> bf16 for the Wo projection
// ---------------------------------------------------------------------------
__global__ void k_gate(const float* __restrict__ attn,
                       const float* __restrict__ gates,
                       unsigned short* __restrict__ gated, int n) {
    int i = blockIdx.x * blockDim.x + threadIdx.x;
    if (i < n) {
        float g = gates[i];
        float s = 1.0f / (1.0f + __expf(-g));
        gated[i] = f32_to_bf16(attn[i] * s);
    }
}

// ---------------------------------------------------------------------------
// out = LayerNorm(x + y) * gamma + beta   (one 128-thread block per row)
// ---------------------------------------------------------------------------
__global__ __launch_bounds__(128)
void k_residual_ln(const float* __restrict__ x, const float* __restrict__ y,
                   const float* __restrict__ gamma, const float* __restrict__ beta,
                   float* __restrict__ out) {
    __shared__ float red[128];
    const int row = blockIdx.x;
    const int tid = threadIdx.x;
    const float* xr = x + (size_t)row * DIMC;
    const float* yr = y + (size_t)row * DIMC;

    float vals[4];
    float s = 0.0f, ss = 0.0f;
#pragma unroll
    for (int i = 0; i < 4; ++i) {
        float v = xr[tid + i * 128] + yr[tid + i * 128];
        vals[i] = v;
        s  += v;
        ss += v * v;
    }
    red[tid] = s;  __syncthreads();
    for (int off = 64; off > 0; off >>= 1) {
        if (tid < off) red[tid] += red[tid + off];
        __syncthreads();
    }
    float mu = red[0] * (1.0f / DIMC);
    __syncthreads();
    red[tid] = ss; __syncthreads();
    for (int off = 64; off > 0; off >>= 1) {
        if (tid < off) red[tid] += red[tid + off];
        __syncthreads();
    }
    float var  = red[0] * (1.0f / DIMC) - mu * mu;
    float rstd = rsqrtf(var + LN_EPS);
#pragma unroll
    for (int i = 0; i < 4; ++i) {
        int c = tid + i * 128;
        out[(size_t)row * DIMC + c] = (vals[i] - mu) * rstd * gamma[c] + beta[c];
    }
}

// ---------------------------------------------------------------------------
// Host-side orchestration
// ---------------------------------------------------------------------------
extern "C" void kernel_launch(void* const* d_in, const int* in_sizes, int n_in,
                              void* d_out, int out_size, void* d_ws, size_t ws_size,
                              hipStream_t stream) {
    (void)in_sizes; (void)n_in; (void)out_size; (void)ws_size;

    const float* x     = (const float*)d_in[0];   // [4,2048,512]
    const float* Wq    = (const float*)d_in[1];   // [512,512]
    const float* Wkv   = (const float*)d_in[2];   // [512,1024]
    const float* Wo    = (const float*)d_in[3];   // [512,512]
    const float* bo    = (const float*)d_in[4];   // [512]
    const float* Wg    = (const float*)d_in[5];   // [512,512]
    const float* bg    = (const float*)d_in[6];   // [512]
    const float* gamma = (const float*)d_in[7];   // [512]
    const float* beta  = (const float*)d_in[8];   // [512]
    float* out = (float*)d_out;

    // Carve the workspace (256-byte aligned slabs).
    char*  w = (char*)d_ws;
    size_t o = 0;
    auto carve = [&](size_t bytes) -> char* {
        size_t r = (o + 255) & ~(size_t)255;
        o = r + bytes;
        return w + r;
    };
    unsigned short* x_bf    = (unsigned short*)carve((size_t)ROWS * DIMC * 2);
    unsigned short* wq_bf   = (unsigned short*)carve((size_t)DIMC * DIMC * 2);
    unsigned short* wkv_bf  = (unsigned short*)carve((size_t)DIMC * 2 * DIMC * 2);
    unsigned short* wg_bf   = (unsigned short*)carve((size_t)DIMC * DIMC * 2);
    unsigned short* wo_bf   = (unsigned short*)carve((size_t)DIMC * DIMC * 2);
    unsigned short* q_bf    = (unsigned short*)carve((size_t)ROWS * DIMC * 2);
    unsigned short* kv_bf   = (unsigned short*)carve((size_t)ROWS * 2 * DIMC * 2);
    float*          gates_f = (float*)carve((size_t)ROWS * DIMC * 4);
    float*          attn_f  = (float*)carve((size_t)ROWS * DIMC * 4);
    unsigned short* gated_b = (unsigned short*)carve((size_t)ROWS * DIMC * 2);
    float*          y_f     = (float*)carve((size_t)ROWS * DIMC * 4);

    const int n_x  = ROWS * DIMC;          // 4,194,304
    const int n_w  = DIMC * DIMC;          //   262,144
    const int n_kv = DIMC * 2 * DIMC;      //   524,288

    // 1) fp32 -> bf16 conversions
    k_cvt_bf16<<<(n_x  + 255) / 256, 256, 0, stream>>>(x,   x_bf,   n_x);
    k_cvt_bf16<<<(n_w  + 255) / 256, 256, 0, stream>>>(Wq,  wq_bf,  n_w);
    k_cvt_bf16<<<(n_kv + 255) / 256, 256, 0, stream>>>(Wkv, wkv_bf, n_kv);
    k_cvt_bf16<<<(n_w  + 255) / 256, 256, 0, stream>>>(Wg,  wg_bf,  n_w);
    k_cvt_bf16<<<(n_w  + 255) / 256, 256, 0, stream>>>(Wo,  wo_bf,  n_w);

    // 2) Projections (WMMA GEMMs)
    const float qscale = 0.125f;           // DIM_HEAD^-0.5
    dim3 blk(128);
    // q = (x @ Wq) * scale  -> bf16
    k_gemm_bf16<<<dim3(DIMC / 64, ROWS / 64), blk, 0, stream>>>(
        x_bf, wq_bf, ROWS, DIMC, DIMC, nullptr, qscale, nullptr, q_bf);
    // kv = x @ Wkv -> bf16
    k_gemm_bf16<<<dim3(2 * DIMC / 64, ROWS / 64), blk, 0, stream>>>(
        x_bf, wkv_bf, ROWS, 2 * DIMC, DIMC, nullptr, 1.0f, nullptr, kv_bf);
    // gates = x @ Wg + bg -> fp32
    k_gemm_bf16<<<dim3(DIMC / 64, ROWS / 64), blk, 0, stream>>>(
        x_bf, wg_bf, ROWS, DIMC, DIMC, bg, 1.0f, gates_f, nullptr);

    // 3) Flash attention (scores never leave the WGP)
    k_flash_attn<<<dim3(SEQ / 64, BATCH * HEADS), blk, 0, stream>>>(
        q_bf, kv_bf, attn_f);

    // 4) Sigmoid gating -> bf16
    k_gate<<<(n_x + 255) / 256, 256, 0, stream>>>(attn_f, gates_f, gated_b, n_x);

    // 5) y = gated @ Wo + bo -> fp32
    k_gemm_bf16<<<dim3(DIMC / 64, ROWS / 64), blk, 0, stream>>>(
        gated_b, wo_bf, ROWS, DIMC, DIMC, bo, 1.0f, y_f, nullptr);

    // 6) out = LayerNorm(x + y) * gamma + beta
    k_residual_ln<<<ROWS, blk, 0, stream>>>(x, y_f, gamma, beta, out);
}
